// CuGraphDistributedGIN_30520037606038
// MI455X (gfx1250) — matrix operand order repaired and checked by reading, not compiled, not run on hardware
//
#include <hip/hip_runtime.h>
#include <hip/hip_bf16.h>

typedef __attribute__((ext_vector_type(16))) _Float16 v16h;
typedef __attribute__((ext_vector_type(8)))  _Float16 v8h;
typedef __attribute__((ext_vector_type(8)))  float    v8f;

#define GIN_N     100000
#define GIN_E     1600000
#define GIN_NG    1024
#define GIN_NPAD  100096   /* 782 * 128 */
#define LDSW      136      /* padded LDS stride in halves: 128 + 8 (bank spread) */

// ---------------------------------------------------------------------------
// Weight prep: W [128, cols] f32 row-major  ->  Wt [cols, 128] f16 row-major
// ---------------------------------------------------------------------------
__global__ __launch_bounds__(256)
void gin_convert_wt(const float* __restrict__ W, _Float16* __restrict__ Wt, int cols) {
    int idx = blockIdx.x * 256 + threadIdx.x;
    if (idx >= 128 * cols) return;
    int k = idx / cols, c = idx % cols;
    Wt[c * 128 + k] = (_Float16)W[idx];
}

// ---------------------------------------------------------------------------
// z[n] = h[n] (pad rows zeroed): the (1+eps)*h term, eps = 0
// ---------------------------------------------------------------------------
__global__ __launch_bounds__(256)
void gin_init_z(const float* __restrict__ h, float* __restrict__ z, int nvalid) {
    int t = blockIdx.x * 256 + threadIdx.x;
    int n = t >> 5;
    if (n >= GIN_NPAD) return;
    int c = (t & 31) * 4;
    float4 v = make_float4(0.f, 0.f, 0.f, 0.f);
    if (n < nvalid) v = *(const float4*)(h + (size_t)n * 128 + c);
    *(float4*)(z + (size_t)n * 128 + c) = v;
}

// ---------------------------------------------------------------------------
// z[dst] += h[src] for each edge (sum aggregation)
// ---------------------------------------------------------------------------
__global__ __launch_bounds__(256)
void gin_scatter_edges(const float* __restrict__ h, const int* __restrict__ ei,
                       float* __restrict__ z) {
    int t = blockIdx.x * 256 + threadIdx.x;
    int e = t >> 5;
    if (e >= GIN_E) return;
    int c = (t & 31) * 4;
    int src = ei[e];
    int dst = ei[GIN_E + e];
    const float4 v = *(const float4*)(h + (size_t)src * 128 + c);
    float* zp = z + (size_t)dst * 128 + c;
    atomicAdd(zp + 0, v.x); atomicAdd(zp + 1, v.y);
    atomicAdd(zp + 2, v.z); atomicAdd(zp + 3, v.w);
}

// ---------------------------------------------------------------------------
// WMMA GEMM: out[rows, ncols] = f(A[rows,128]) @ Wt^T + bias
//   Wt is [ncols, 128] f16 (pre-transposed), staged in LDS.
//   inTransform: A elt -> relu(a*scale[k] + shift[k])  (fused BN+ReLU)
//   outRelu:     output relu
// Block = 256 threads = 8 waves; block computes 128 rows; wave computes 16 rows.
// ---------------------------------------------------------------------------
__global__ __launch_bounds__(256)
void gin_gemm_wmma(const float* __restrict__ A, const _Float16* __restrict__ Wt,
                   const float* __restrict__ bias, float* __restrict__ out,
                   int ncols,
                   const float* __restrict__ scale, const float* __restrict__ shift,
                   int inTransform, int outRelu) {
    __shared__ _Float16 sW[128 * LDSW];

    const int tid = threadIdx.x;
    // stage Wt into LDS (uint4 = 8 halves per copy), padded row stride
    const int nq = ncols * 16;
    for (int q = tid; q < nq; q += 256) {
        int col = q >> 4, kq = q & 15;
        *(uint4*)(&sW[col * LDSW + kq * 8]) = ((const uint4*)Wt)[q];
    }
    __syncthreads();

    const int wave  = tid >> 5;
    const int lane  = tid & 31;
    const int mrow  = lane & 15;
    const int khalf = lane >> 4;                // 0 or 1
    const int rowBase = blockIdx.x * 128 + wave * 16;
    const float* arow = A + (size_t)(rowBase + mrow) * 128;

    // A fragments for the 4 K-steps (16x32 f16 layout per CDNA5 ISA)
    v16h afrag[4];
#pragma unroll
    for (int kk = 0; kk < 4; ++kk) {
        int k0 = kk * 32 + khalf * 8;           // elems 0..7 : K = k0..k0+7
        int k1 = k0 + 16;                       // elems 8..15: K = k1..k1+7
        v8f f0 = *(const v8f*)(arow + k0);
        v8f f1 = *(const v8f*)(arow + k1);
        if (inTransform) {
#pragma unroll
            for (int j = 0; j < 8; ++j) {
                float t0 = f0[j] * scale[k0 + j] + shift[k0 + j];
                float t1 = f1[j] * scale[k1 + j] + shift[k1 + j];
                f0[j] = t0 > 0.f ? t0 : 0.f;
                f1[j] = t1 > 0.f ? t1 : 0.f;
            }
        }
        v8h h0 = __builtin_convertvector(f0, v8h);
        v8h h1 = __builtin_convertvector(f1, v8h);
        v16h av;
#pragma unroll
        for (int j = 0; j < 8; ++j) { av[j] = h0[j]; av[j + 8] = h1[j]; }
        afrag[kk] = av;
    }

    const int nct = ncols >> 4;
    for (int ct = 0; ct < nct; ++ct) {
        const int col = ct * 16 + mrow;         // this lane's B column / C column
        v8f acc = {};
#pragma unroll
        for (int kk = 0; kk < 4; ++kk) {
            int koff = kk * 32 + khalf * 8;     // elems 0..7: K koff..koff+7 ; elems 8..15: K koff+16..koff+23
            v8h b0 = *(const v8h*)(&sW[col * LDSW + koff]);
            v8h b1 = *(const v8h*)(&sW[col * LDSW + koff + 16]);
            v16h bv;
#pragma unroll
            for (int j = 0; j < 8; ++j) { bv[j] = b0[j]; bv[j + 8] = b1[j]; }
            acc = __builtin_amdgcn_wmma_f32_16x16x32_f16(
                false, afrag[kk], false, bv, (short)0, acc, false, false);
        }
        const float bc = bias[col];
        float* op = out + (size_t)(rowBase + khalf * 8) * ncols + col;
#pragma unroll
        for (int j = 0; j < 8; ++j) {           // C layout: vgpr j -> row khalf*8 + j
            float v = acc[j] + bc;
            if (outRelu) v = v > 0.f ? v : 0.f;
            op[(size_t)j * ncols] = v;
        }
    }
}

// ---------------------------------------------------------------------------
// Per-column sum / sum-of-squares over the first N rows of Y[*,128]
// ---------------------------------------------------------------------------
__global__ __launch_bounds__(256)
void gin_col_stats(const float* __restrict__ Y, float* __restrict__ sum,
                   float* __restrict__ sq, int nrows) {
    __shared__ float s1[256], s2[256];
    const int col  = threadIdx.x & 127;
    const int half = threadIdx.x >> 7;
    int r    = blockIdx.x * 256 + half;
    int rend = blockIdx.x * 256 + 256;
    if (rend > nrows) rend = nrows;
    float a = 0.f, b = 0.f;
    for (; r < rend; r += 2) {
        float v = Y[(size_t)r * 128 + col];
        a += v; b += v * v;
    }
    s1[threadIdx.x] = a; s2[threadIdx.x] = b;
    __syncthreads();
    if (threadIdx.x < 128) {
        atomicAdd(&sum[col], s1[threadIdx.x] + s1[threadIdx.x + 128]);
        atomicAdd(&sq[col],  s2[threadIdx.x] + s2[threadIdx.x + 128]);
    }
}

__global__ void gin_bn_prep(const float* __restrict__ sum, const float* __restrict__ sq,
                            const float* __restrict__ gamma, const float* __restrict__ beta,
                            float* __restrict__ scale, float* __restrict__ shift) {
    int c = threadIdx.x;
    if (c < 128) {
        float inv = 1.0f / (float)GIN_N;
        float mu  = sum[c] * inv;
        float var = sq[c] * inv - mu * mu;
        float rs  = rsqrtf(var + 1e-5f);
        float sc  = gamma[c] * rs;
        scale[c] = sc;
        shift[c] = beta[c] - mu * sc;
    }
}

__global__ __launch_bounds__(256)
void gin_zero_f32(float* __restrict__ p, int n) {
    int t = blockIdx.x * 256 + threadIdx.x;
    if (t < n) p[t] = 0.f;
}

// ---------------------------------------------------------------------------
// g[batch[n]] += h[n]  (global add pool)
// ---------------------------------------------------------------------------
__global__ __launch_bounds__(256)
void gin_pool(const float* __restrict__ h, const int* __restrict__ batch,
              float* __restrict__ g) {
    int t = blockIdx.x * 256 + threadIdx.x;
    int n = t >> 5;
    if (n >= GIN_N) return;
    int c = (t & 31) * 4;
    int b = batch[n];
    const float4 v = *(const float4*)(h + (size_t)n * 128 + c);
    float* gp = g + (size_t)b * 128 + c;
    atomicAdd(gp + 0, v.x); atomicAdd(gp + 1, v.y);
    atomicAdd(gp + 2, v.z); atomicAdd(gp + 3, v.w);
}

// ---------------------------------------------------------------------------
extern "C" void kernel_launch(void* const* d_in, const int* in_sizes, int n_in,
                              void* d_out, int out_size, void* d_ws, size_t ws_size,
                              hipStream_t stream) {
    (void)in_sizes; (void)n_in; (void)out_size; (void)ws_size;

    // ---- inputs (jax pytree flatten: sorted dict keys; "cls" < "convs") ----
    const float* x     = (const float*)d_in[0];
    const int*   ei    = (const int*)d_in[1];
    const int*   batch = (const int*)d_in[2];
    const float* clsW1 = (const float*)d_in[3];   // cls: W1, W2, b1, b2
    const float* clsW2 = (const float*)d_in[4];
    const float* clsb1 = (const float*)d_in[5];
    const float* clsb2 = (const float*)d_in[6];
    const float* cW1[3], *cW2[3], *cb1[3], *cb2[3], *cBeta[3], *cGamma[3];
    for (int i = 0; i < 3; ++i) {                 // convs[i]: W1,W2,b1,b2,beta,gamma
        cW1[i]    = (const float*)d_in[7  + 6 * i];
        cW2[i]    = (const float*)d_in[8  + 6 * i];
        cb1[i]    = (const float*)d_in[9  + 6 * i];
        cb2[i]    = (const float*)d_in[10 + 6 * i];
        cBeta[i]  = (const float*)d_in[11 + 6 * i];
        cGamma[i] = (const float*)d_in[12 + 6 * i];
    }

    // ---- workspace carve-up ----
    char* ws = (char*)d_ws;
    size_t off = 0;
    auto carve = [&](size_t bytes) { char* p = ws + off; off += (bytes + 255) & ~(size_t)255; return p; };
    float*     buf0  = (float*)    carve((size_t)GIN_NPAD * 128 * 4);
    float*     buf1  = (float*)    carve((size_t)GIN_NPAD * 128 * 4);
    _Float16*  wt    = (_Float16*) carve((size_t)8 * 16384 * 2);   // 8 slots of [<=128,128] f16
    float*     ssum  = (float*)    carve(128 * 4);
    float*     ssq   = (float*)    carve(128 * 4);
    float*     scale = (float*)    carve(128 * 4);
    float*     shift = (float*)    carve(128 * 4);
    float*     g     = (float*)    carve((size_t)GIN_NG * 128 * 4);
    float*     tcls  = (float*)    carve((size_t)GIN_NG * 128 * 4);

    // ---- weight prep: transpose + f16 convert ----
    for (int i = 0; i < 3; ++i) {
        gin_convert_wt<<<64, 256, 0, stream>>>(cW1[i], wt + (2 * i)     * 16384, 128);
        gin_convert_wt<<<64, 256, 0, stream>>>(cW2[i], wt + (2 * i + 1) * 16384, 128);
    }
    gin_convert_wt<<<64, 256, 0, stream>>>(clsW1, wt + 6 * 16384, 128);
    gin_convert_wt<<<32, 256, 0, stream>>>(clsW2, wt + 7 * 16384, 64);

    const int gemmGrid    = GIN_NPAD / 128;            // 782
    const int initGrid    = GIN_NPAD * 32 / 256;       // 12512
    const int scatterGrid = GIN_E * 32 / 256;          // 200000
    const int statsGrid   = (GIN_N + 255) / 256;       // 391

    // ---- 3 GIN conv layers ----
    const float* hin = x;
    for (int i = 0; i < 3; ++i) {
        float* zb   = (i & 1) ? buf1 : buf0;
        float* Yb   = (i & 1) ? buf0 : buf1;
        float* hout = zb;                               // z dead after GEMM1

        gin_init_z<<<initGrid, 256, 0, stream>>>(hin, zb, GIN_N);
        gin_scatter_edges<<<scatterGrid, 256, 0, stream>>>(hin, ei, zb);
        // Y = z @ W1 + b1
        gin_gemm_wmma<<<gemmGrid, 256, 0, stream>>>(zb, wt + (2 * i) * 16384, cb1[i],
                                                    Yb, 128, nullptr, nullptr, 0, 0);
        // batchnorm stats + fold
        gin_zero_f32<<<1, 256, 0, stream>>>(ssum, 256);  // ssum & ssq are contiguous carves: zeroes both
        gin_zero_f32<<<1, 256, 0, stream>>>(ssq, 128);   // zero ssq explicitly too
        gin_col_stats<<<statsGrid, 256, 0, stream>>>(Yb, ssum, ssq, GIN_N);
        gin_bn_prep<<<1, 128, 0, stream>>>(ssum, ssq, cGamma[i], cBeta[i], scale, shift);
        // h = relu( relu(bn(Y)) @ W2 + b2 )
        gin_gemm_wmma<<<gemmGrid, 256, 0, stream>>>(Yb, wt + (2 * i + 1) * 16384, cb2[i],
                                                    hout, 128, scale, shift, 1, 1);
        hin = hout;
    }

    // ---- global add pool + classifier ----
    gin_zero_f32<<<(GIN_NG * 128) / 256, 256, 0, stream>>>(g, GIN_NG * 128);
    gin_pool<<<GIN_N * 32 / 256, 256, 0, stream>>>(hin, batch, g);
    gin_gemm_wmma<<<GIN_NG / 128, 256, 0, stream>>>(g, wt + 6 * 16384, clsb1,
                                                    tcls, 128, nullptr, nullptr, 0, 1);
    gin_gemm_wmma<<<GIN_NG / 128, 256, 0, stream>>>(tcls, wt + 7 * 16384, clsb2,
                                                    (float*)d_out, 64, nullptr, nullptr, 0, 0);
}
